// AttentionNet_57767310131468
// MI455X (gfx1250) — matrix operand order repaired and checked by reading, not compiled, not run on hardware
//
#include <hip/hip_runtime.h>
#include <hip/hip_bf16.h>

typedef __attribute__((ext_vector_type(16))) _Float16 v16h;
typedef __attribute__((ext_vector_type(8)))  _Float16 v8h;
typedef __attribute__((ext_vector_type(8)))  float    v8f;

#define Bsz 128
#define Lsz 512
#define Dsz 300
#define Hsz 200
#define KP  320   // D padded to 10 chunks of K=32
#define HP  208   // H padded to 13 tiles of N=16

// Fast tanh on the hardware transcendental pipe: v_exp_f32 + v_rcp_f32.
// |err| ~1e-6, far below the f16 input quantization already applied.
__device__ __forceinline__ float fast_tanh(float x) {
    x = fminf(fmaxf(x, -20.0f), 20.0f);          // keep e^{2x} finite
    float e2 = __expf(2.0f * x);                  // v_exp_f32
    return (e2 - 1.0f) * __builtin_amdgcn_rcpf(e2 + 1.0f);  // v_rcp_f32
}

// ---------------------------------------------------------------------------
// Prep: w1t_pad[n][k] = w1[k][n] (f16, zero-padded), w2pad[n] (f32, padded)
// ---------------------------------------------------------------------------
__global__ __launch_bounds__(256)
void prep_w_kernel(const float* __restrict__ w1, const float* __restrict__ w2,
                   _Float16* __restrict__ w1t, float* __restrict__ w2pad) {
    int idx = blockIdx.x * 256 + threadIdx.x;
    if (idx < HP * KP) {
        int n = idx / KP, k = idx % KP;
        float v = (n < Hsz && k < Dsz) ? w1[k * Hsz + n] : 0.0f;
        w1t[idx] = (_Float16)v;
    }
    if (idx < HP) w2pad[idx] = (idx < Hsz) ? w2[idx] : 0.0f;
}

// ---------------------------------------------------------------------------
// Gather embeddings -> f16 padded [B][L][KP], masked max pools -> subj/obj emb
// ---------------------------------------------------------------------------
__global__ __launch_bounds__(KP)
void pool_gather_kernel(const int* __restrict__ words, const int* __restrict__ subj,
                        const int* __restrict__ obj, const float* __restrict__ emb,
                        _Float16* __restrict__ wordemb, float* __restrict__ subj_emb,
                        float* __restrict__ obj_emb) {
    __shared__ int sw[Lsz];
    __shared__ unsigned char sm[Lsz], om[Lsz];
    int b = blockIdx.x, t = threadIdx.x;
    for (int i = t; i < Lsz; i += KP) {
        sw[i] = words[b * Lsz + i];
        sm[i] = (subj[b * Lsz + i] == 0);
        om[i] = (obj[b * Lsz + i] == 0);
    }
    __syncthreads();
    if (t < Dsz) {
        float smax = -1e12f, omax = -1e12f;
        for (int l = 0; l < Lsz; ++l) {
            float v = emb[(size_t)sw[l] * Dsz + t];
            wordemb[((size_t)b * Lsz + l) * KP + t] = (_Float16)v;
            if (sm[l]) smax = fmaxf(smax, v);
            if (om[l]) omax = fmaxf(omax, v);
        }
        subj_emb[b * Dsz + t] = smax;
        obj_emb[b * Dsz + t] = omax;
    } else {
        for (int l = 0; l < Lsz; ++l)
            wordemb[((size_t)b * Lsz + l) * KP + t] = (_Float16)0.0f;
    }
}

// ---------------------------------------------------------------------------
// sup_proj[b][h] = subj_emb[b] @ w1[D:2D, h] + b1[h]   (padded to HP)
// ---------------------------------------------------------------------------
__global__ __launch_bounds__(256)
void sup_proj_kernel(const float* __restrict__ subj_emb, const float* __restrict__ w1,
                     const float* __restrict__ b1, float* __restrict__ supproj) {
    __shared__ float se[Dsz];
    int b = blockIdx.x, t = threadIdx.x;
    for (int i = t; i < Dsz; i += 256) se[i] = subj_emb[b * Dsz + i];
    __syncthreads();
    if (t < HP) {
        float acc = 0.0f;
        if (t < Hsz) {
            acc = b1[t];
            for (int d = 0; d < Dsz; ++d)
                acc += se[d] * w1[(Dsz + d) * Hsz + t];
        }
        supproj[b * HP + t] = acc;
    }
}

// ---------------------------------------------------------------------------
// Fused score GEMM (WMMA f16 -> f32): scores[m] = w2 . tanh(A@W1t + supproj)
// One wave per 16-row M-tile; loops 13 N-tiles x 10 K-chunks of 16x16x32 WMMA.
// ---------------------------------------------------------------------------
__global__ __launch_bounds__(256)
void score_gemm_kernel(const _Float16* __restrict__ A,    // [B*L][KP]
                       const _Float16* __restrict__ Wt,   // [HP][KP]
                       const float* __restrict__ supproj, // [B][HP]
                       const float* __restrict__ w2pad,   // [HP]
                       float* __restrict__ scores) {      // [B*L]
    const int lane  = threadIdx.x & 31;
    const int wave  = threadIdx.x >> 5;
    const int mbase = (blockIdx.x * 8 + wave) * 16;
    const int b     = mbase >> 9;            // L == 512
    const int row   = lane & 15;
    const int half  = lane >> 4;

    // Hoist the 10 A fragments (ISA 16-bit A layout: half 0 -> K{0-7,16-23})
    v16h afrag[10];
    const _Float16* arow = A + (size_t)(mbase + row) * KP + half * 8;
#pragma unroll
    for (int kt = 0; kt < 10; ++kt) {
        v8h lo = *(const v8h*)(arow + kt * 32);
        v8h hi = *(const v8h*)(arow + kt * 32 + 16);
        afrag[kt] = __builtin_shufflevector(lo, hi, 0,1,2,3,4,5,6,7,8,9,10,11,12,13,14,15);
    }

    float part[8];
#pragma unroll
    for (int j = 0; j < 8; ++j) part[j] = 0.0f;

    for (int nt = 0; nt < 13; ++nt) {
        const int ncol = nt * 16 + row;           // this lane's output column
        const float sp  = supproj[b * HP + ncol];
        const float w2v = w2pad[ncol];
        // B layout: lane holds column `row`, half 0 -> K{0-7,16-23} (matches A)
        const _Float16* brow = Wt + (size_t)ncol * KP + half * 8;
        v8f c = {};
#pragma unroll
        for (int kt = 0; kt < 10; ++kt) {
            v8h lo = *(const v8h*)(brow + kt * 32);
            v8h hi = *(const v8h*)(brow + kt * 32 + 16);
            v16h bfrag = __builtin_shufflevector(lo, hi, 0,1,2,3,4,5,6,7,8,9,10,11,12,13,14,15);
            c = __builtin_amdgcn_wmma_f32_16x16x32_f16(
                    false, afrag[kt], false, bfrag, (short)0, c, false, false);
        }
#pragma unroll
        for (int j = 0; j < 8; ++j) {
            float hid = fast_tanh(c[j] + sp);     // branch-free transcendental path
            part[j] += hid * w2v;
        }
    }
    // Reduce the 16-lane N dimension inside each half-wave
#pragma unroll
    for (int j = 0; j < 8; ++j) {
        float v = part[j];
        v += __shfl_xor(v, 1, 32);
        v += __shfl_xor(v, 2, 32);
        v += __shfl_xor(v, 4, 32);
        v += __shfl_xor(v, 8, 32);
        part[j] = v;
    }
    if (row == 0) {
#pragma unroll
        for (int j = 0; j < 8; ++j)
            scores[mbase + half * 8 + j] = part[j];   // lanes 0..15: M=j; 16..31: M=8+j
    }
}

// ---------------------------------------------------------------------------
// Softmax over L + attention-weighted sum of embeddings -> attn_out [B][D]
// ---------------------------------------------------------------------------
__global__ __launch_bounds__(Lsz)
void softmax_attn_kernel(const float* __restrict__ scores, const int* __restrict__ words,
                         const float* __restrict__ emb, float* __restrict__ attn_out) {
    __shared__ float sattn[Lsz];
    __shared__ float red[Lsz];
    __shared__ int   sw[Lsz];
    int b = blockIdx.x, t = threadIdx.x;
    float s = scores[b * Lsz + t];
    sw[t] = words[b * Lsz + t];
    red[t] = s;
    __syncthreads();
    for (int st = Lsz / 2; st > 0; st >>= 1) {
        if (t < st) red[t] = fmaxf(red[t], red[t + st]);
        __syncthreads();
    }
    float mx = red[0];
    __syncthreads();
    float e = expf(s - mx);
    red[t] = e;
    __syncthreads();
    for (int st = Lsz / 2; st > 0; st >>= 1) {
        if (t < st) red[t] += red[t + st];
        __syncthreads();
    }
    float inv = 1.0f / red[0];
    __syncthreads();
    sattn[t] = e * inv;
    __syncthreads();
    if (t < Dsz) {
        float acc = 0.0f;
        for (int l = 0; l < Lsz; ++l)
            acc += sattn[l] * emb[(size_t)sw[l] * Dsz + t];
        attn_out[b * Dsz + t] = acc;
    }
}

// ---------------------------------------------------------------------------
// Final MLP: relu(cat @ mw1 + mb1) @ mw2 + mb2, relu.  cat = [attn,attn,s,o]
// ---------------------------------------------------------------------------
__global__ __launch_bounds__(256)
void mlp_kernel(const float* __restrict__ attn, const float* __restrict__ subj_emb,
                const float* __restrict__ obj_emb, const float* __restrict__ mw1,
                const float* __restrict__ mb1, const float* __restrict__ mw2,
                const float* __restrict__ mb2, float* __restrict__ out) {
    __shared__ float cat[4 * Dsz];
    __shared__ float h1[Hsz];
    int b = blockIdx.x, t = threadIdx.x;
    for (int i = t; i < 4 * Dsz; i += 256) {
        float v;
        if (i < Dsz)            v = attn[b * Dsz + i];
        else if (i < 2 * Dsz)   v = attn[b * Dsz + i - Dsz];       // obj_attn == subj_attn (source bug)
        else if (i < 3 * Dsz)   v = subj_emb[b * Dsz + i - 2 * Dsz];
        else                    v = obj_emb[b * Dsz + i - 3 * Dsz];
        cat[i] = v;
    }
    __syncthreads();
    if (t < Hsz) {
        float acc = mb1[t];
        for (int k = 0; k < 4 * Dsz; ++k) acc += cat[k] * mw1[k * Hsz + t];
        h1[t] = fmaxf(acc, 0.0f);
    }
    __syncthreads();
    if (t < Hsz) {
        float acc = mb2[t];
        for (int k = 0; k < Hsz; ++k) acc += h1[k] * mw2[k * Hsz + t];
        out[b * Hsz + t] = fmaxf(acc, 0.0f);
    }
}

// ---------------------------------------------------------------------------
extern "C" void kernel_launch(void* const* d_in, const int* in_sizes, int n_in,
                              void* d_out, int out_size, void* d_ws, size_t ws_size,
                              hipStream_t stream) {
    const int*   words = (const int*)d_in[0];
    const int*   subj  = (const int*)d_in[1];
    const int*   obj   = (const int*)d_in[2];
    const float* emb   = (const float*)d_in[3];
    const float* w1    = (const float*)d_in[4];
    const float* b1    = (const float*)d_in[5];
    const float* w2    = (const float*)d_in[6];
    // d_in[7] = b2: uniform shift of scores, softmax-invariant -> unused
    const float* mw1   = (const float*)d_in[8];
    const float* mb1   = (const float*)d_in[9];
    const float* mw2   = (const float*)d_in[10];
    const float* mb2   = (const float*)d_in[11];
    float* out = (float*)d_out;

    char* ws = (char*)d_ws;
    size_t off = 0;
    auto alloc = [&](size_t bytes) -> void* {
        void* p = ws + off;
        off = (off + bytes + 255) & ~(size_t)255;
        return p;
    };
    _Float16* wordemb = (_Float16*)alloc((size_t)Bsz * Lsz * KP * sizeof(_Float16));
    _Float16* w1t     = (_Float16*)alloc((size_t)HP * KP * sizeof(_Float16));
    float* supproj    = (float*)alloc((size_t)Bsz * HP * sizeof(float));
    float* w2pad      = (float*)alloc((size_t)HP * sizeof(float));
    float* subj_emb   = (float*)alloc((size_t)Bsz * Dsz * sizeof(float));
    float* obj_emb    = (float*)alloc((size_t)Bsz * Dsz * sizeof(float));
    float* scores     = (float*)alloc((size_t)Bsz * Lsz * sizeof(float));
    float* attn_out   = (float*)alloc((size_t)Bsz * Dsz * sizeof(float));

    prep_w_kernel<<<(HP * KP + 255) / 256, 256, 0, stream>>>(w1, w2, w1t, w2pad);
    pool_gather_kernel<<<Bsz, KP, 0, stream>>>(words, subj, obj, emb,
                                               wordemb, subj_emb, obj_emb);
    sup_proj_kernel<<<Bsz, 256, 0, stream>>>(subj_emb, w1, b1, supproj);
    // B*L/16 = 4096 M-tiles, 8 waves per 256-thread block -> 512 blocks
    score_gemm_kernel<<<(Bsz * Lsz / 16) / 8, 256, 0, stream>>>(wordemb, w1t, supproj,
                                                                w2pad, scores);
    softmax_attn_kernel<<<Bsz, Lsz, 0, stream>>>(scores, words, emb, attn_out);
    mlp_kernel<<<Bsz, 256, 0, stream>>>(attn_out, subj_emb, obj_emb,
                                        mw1, mb1, mw2, mb2, out);
}
